// SpMiddleResNetFHDELKv3_64613488001619
// MI455X (gfx1250) — matrix operand order repaired and checked by reading, not compiled, not run on hardware
//
// SpMiddleResNetFHDELKv3 for MI455X (gfx1250): implicit-GEMM sparse-3D-conv
// pipeline built on v_wmma_f32_16x16x32_bf16 (wave32).
//
// Rationale (compile-only; reasoned from MI455X specs):
//  * Total math ~65 GFLOP; working set (bf16 activations + weights) ~70 MB
//    << 192 MB global L2 -> L2-resident; HBM (23.3 TB/s) touched ~once.
//  * bf16 A/B with f32 accumulate: K=32 per WMMA (8x the K-depth of the f32
//    16x16x4 op), fp32-compatible dynamic range, half the activation bytes.
//  * Grid maps to (x-tile, y-quad, z): z,y are wave-uniform (scalar tap
//    bounds checks, zero div/mod), A rows are x-contiguous in NDHWC ->
//    coalesced 16x16B group loads.
//  * Weight slab per (tap,kchunk) (<=8KB) staged into LDS once per block via
//    global_load_async_to_lds_b128 (ASYNCcnt + s_wait_asynccnt), ds_store
//    fallback otherwise; B fragments then read from LDS (ds_load), cutting
//    the 4x per-wave global weight traffic and using the idle 320KB/WGP LDS.
//    Next slab is prefetched (global_prefetch_b8).
//  * BN + bias + mask + residual + ReLU fused into the WMMA epilogue
//    (mask in {0,1} distributes through the affine BN).
//
// Input flattening assumption: setup_inputs() dict insertion order, recursing
// in insertion order: 0 voxel_features, 1 coors, 2 batch_size, 3 input_shape,
// 4.. params: stem{w,bn(g,b,m,v)}, levels[i]{(down{w,bn})?, blocks[2]{w1,b1,
// bn1(4),w2,b2,bn2(4)}, tail{w,bn}, elk_tail{w,bn}}, extra{w,bn}.

#include <hip/hip_runtime.h>
#include <hip/hip_bf16.h>
#include <cstdint>

#define EPS_BN 0.001f
#define AS1 __attribute__((address_space(1)))
#define AS3 __attribute__((address_space(3)))

typedef __attribute__((ext_vector_type(16))) __bf16 v16bf;
typedef __attribute__((ext_vector_type(8)))  __bf16 v8bf;
typedef __attribute__((ext_vector_type(8)))  float  v8f;
typedef __attribute__((ext_vector_type(4)))  int    v4i;

union AB16 { v16bf v; v8bf h[2]; };
union H8   { v8bf v; unsigned long long u[2]; };

__device__ __forceinline__ v8bf zero8() {
  H8 z; z.u[0] = 0ULL; z.u[1] = 0ULL; return z.v;
}

#if __has_builtin(__builtin_amdgcn_global_load_async_to_lds_b128)
#define USE_ASYNC_LDS 1
#else
#define USE_ASYNC_LDS 0
#endif

// Stage `chunks` 16-byte chunks from global `src` into LDS `sh`, cooperatively
// across the 128-thread block. Caller brackets with __syncthreads().
__device__ __forceinline__ void stage_slab(__bf16* sh, const __bf16* src, int chunks) {
#if USE_ASYNC_LDS
  for (int i = threadIdx.x; i < chunks; i += 128) {
    __builtin_amdgcn_global_load_async_to_lds_b128(
        (AS1 v4i*)(src + (long)i * 8), (AS3 v4i*)(sh + (long)i * 8), 0, 0);
  }
#if __has_builtin(__builtin_amdgcn_s_wait_asynccnt)
  __builtin_amdgcn_s_wait_asynccnt(0);
#else
  asm volatile("s_wait_asynccnt 0" ::: "memory");
#endif
#else
  const uint4* s4 = (const uint4*)src;
  uint4* d4 = (uint4*)sh;
  for (int i = threadIdx.x; i < chunks; i += 128) d4[i] = s4[i];
#endif
}

// ---------------------------------------------------------------- utilities

__global__ void fill_u32_kernel(unsigned int* __restrict__ p, unsigned int v, long n) {
  long i = (long)blockIdx.x * blockDim.x + threadIdx.x;
  if (i < n) p[i] = v;
}

__global__ void scatter_kernel(const float* __restrict__ feats,
                               const int* __restrict__ coors,
                               __bf16* __restrict__ dense, float* __restrict__ mask,
                               int n, int H, int W, int Cpad, int Cin) {
  int i = blockIdx.x * blockDim.x + threadIdx.x;
  if (i >= n) return;
  int z = coors[i * 4 + 1], y = coors[i * 4 + 2], x = coors[i * 4 + 3];
  long v = ((long)z * H + y) * W + x;
  for (int c = 0; c < Cin; ++c)
    dense[v * Cpad + c] = (__bf16)feats[(long)i * Cin + c];
  mask[v] = 1.0f;
}

// Repack OIDHW fp32 weights into bf16 B-fragment layout:
//   packed[((t*KC + kc)*2 + khalf)*Cout + co][16] holds K = kc*32+khalf*16+kk
__global__ void repack_kernel(const float* __restrict__ w, __bf16* __restrict__ packed,
                              int Cout, int CinW, int taps, int KC) {
  long total = (long)taps * KC * 2 * Cout * 16;
  long i = (long)blockIdx.x * blockDim.x + threadIdx.x;
  if (i >= total) return;
  int  kk = (int)(i & 15);
  long r  = i >> 4;
  int  co = (int)(r % Cout); r /= Cout;
  int  kh = (int)(r & 1);    r >>= 1;
  int  kc = (int)(r % KC);   r /= KC;
  int  t  = (int)r;
  int  ci = kc * 32 + kh * 16 + kk;
  float val = (ci < CinW) ? w[((long)co * CinW + ci) * taps + t] : 0.0f;
  packed[i] = (__bf16)val;
}

__global__ void mask_down_kernel(const float* __restrict__ mi, float* __restrict__ mo,
                                 int Din, int Hin, int Win, int Dout, int Hout, int Wout,
                                 int kd, int kh, int kw, int sz, int sy, int sx,
                                 int pz, int py, int px) {
  long Nout = (long)Dout * Hout * Wout;
  long i = (long)blockIdx.x * blockDim.x + threadIdx.x;
  if (i >= Nout) return;
  int x = (int)(i % Wout);
  int y = (int)((i / Wout) % Hout);
  int z = (int)(i / ((long)Wout * Hout));
  float any = 0.0f;
  for (int dz = 0; dz < kd; ++dz) {
    int iz = z * sz - pz + dz; if (iz < 0 || iz >= Din) continue;
    for (int dy = 0; dy < kh; ++dy) {
      int iy = y * sy - py + dy; if (iy < 0 || iy >= Hin) continue;
      for (int dx = 0; dx < kw; ++dx) {
        int ix = x * sx - px + dx; if (ix < 0 || ix >= Win) continue;
        if (mi[((long)iz * Hin + iy) * Win + ix] > 0.0f) any = 1.0f;
      }
    }
  }
  mo[i] = any;
}

// ex (NDHWC bf16, D=2,H=W=12,C=128) -> out fp32 (C*D, H, W)
__global__ void final_kernel(const __bf16* __restrict__ ex, float* __restrict__ out,
                             int Dd, int Hh, int Ww, int Cc) {
  int total = Cc * Dd * Hh * Ww;
  int i = blockIdx.x * blockDim.x + threadIdx.x;
  if (i >= total) return;
  int w = i % Ww;
  int h = (i / Ww) % Hh;
  int d = (i / (Ww * Hh)) % Dd;
  int c = i / (Ww * Hh * Dd);
  out[i] = (float)ex[(((long)d * Hh + h) * Ww + w) * Cc + c];
}

// --------------------------------------------------- implicit-GEMM conv+WMMA
// Block = 128 threads = 4 waves. blockIdx = (x-tile of 16, y-quad, z); wave w
// owns output row (z, y0+w, x0..x0+15). Each wave accumulates NTILES 16x16
// f32 tiles over K = taps x KC x 32 bf16 channels. Weights come from an
// LDS-staged slab shared by all 4 waves.
template <int NTILES, int KC>
__global__ __launch_bounds__(128)
void conv_wmma_kernel(const __bf16* __restrict__ in, const __bf16* __restrict__ wp,
                      const float* __restrict__ mask, const __bf16* res,
                      const float* __restrict__ bng, const float* __restrict__ bnb,
                      const float* __restrict__ bnm, const float* __restrict__ bnv,
                      const float* bias, __bf16* out,
                      int Din, int Hin, int Win, int Dout, int Hout, int Wout,
                      int Cin, int Cout, int kd, int kh, int kw,
                      int sz, int sy, int sx, int pz, int py, int px,
                      int relu, int addres) {
  __shared__ __align__(16) __bf16 shB[4096];   // one (t,kc) slab: 2*Cout*16 bf16

  const int lane = threadIdx.x & 31;
  const int wv   = threadIdx.x >> 5;
  const int mrow = lane & 15;   // A row (x offset) / B,C,D column in tile
  const int grp  = lane >> 4;   // K-half / M-half selector

  const int x0 = blockIdx.x * 16;
  const int yo = blockIdx.y * 4 + wv;
  const int zo = blockIdx.z;
  const int xo = x0 + mrow;
  const bool rowvalid = (xo < Wout) && (yo < Hout);

  v8f acc[NTILES] = {};

  const int taps = kd * kh * kw;
  const int slabElems  = 2 * Cout * 16;   // bf16 per (t,kc) slab
  const int slabChunks = slabElems / 8;   // 16B chunks

  for (int t = 0; t < taps; ++t) {
    const int dz = t / (kh * kw);
    const int rr = t % (kh * kw);
    const int dy = rr / kw;
    const int dx = rr % kw;
    const int iz = zo * sz - pz + dz;           // wave-uniform
    const int iy = yo * sy - py + dy;           // wave-uniform
    const int ix = xo * sx - px + dx;           // per-lane (x only)
    const bool valid = rowvalid && iz >= 0 && iz < Din && iy >= 0 && iy < Hin &&
                       ix >= 0 && ix < Win;
    const long ibase = valid ? ((long)(iz * Hin + iy) * Win + ix) * Cin : 0;
#pragma unroll
    for (int kc = 0; kc < KC; ++kc) {
      // ---- stage weight slab (t,kc) into LDS, shared by all 4 waves ----
      const __bf16* src = wp + (long)(t * KC + kc) * slabElems;
      __syncthreads();                      // previous slab fully consumed
      stage_slab(shB, src, slabChunks);
      __builtin_prefetch(src + slabElems, 0, 1);  // next slab -> cache
      __syncthreads();                      // slab visible to all waves

      // ---- A fragment (16x32 bf16): group g holds K = g*16 .. g*16+15 ----
      AB16 a;
      const int c0 = kc * 32 + grp * 16;
      const int c1 = c0 + 8;
      a.h[0] = (valid && c0 < Cin) ? *(const v8bf*)(in + ibase + c0) : zero8();
      a.h[1] = (valid && c1 < Cin) ? *(const v8bf*)(in + ibase + c1) : zero8();

#pragma unroll
      for (int nt = 0; nt < NTILES; ++nt) {
        const v16bf b = *(const v16bf*)(shB + (grp * Cout + nt * 16 + mrow) * 16);
        acc[nt] = __builtin_amdgcn_wmma_f32_16x16x32_bf16(
            false, a.v, false, b, (short)0, acc[nt], false, false);
      }
    }
  }

  // Epilogue. C/D layout: VGPR r, lane l -> M = r + 8*(l>>4), N = l&15.
  const long rowbase = ((long)zo * Hout + yo) * Wout;
  float mk[8];
#pragma unroll
  for (int r = 0; r < 8; ++r) {
    const int xr = x0 + grp * 8 + r;
    mk[r] = (xr < Wout && yo < Hout) ? mask[rowbase + xr] : 0.0f;
  }
#pragma unroll
  for (int nt = 0; nt < NTILES; ++nt) {
    const int ch = nt * 16 + mrow;
    const float sc = bng[ch] * rsqrtf(bnv[ch] + EPS_BN);
    float sh = bnb[ch] - bnm[ch] * sc;
    if (bias) sh += bias[ch] * sc;
#pragma unroll
    for (int r = 0; r < 8; ++r) {
      const int xr = x0 + grp * 8 + r;
      if (xr < Wout && yo < Hout) {
        float val = acc[nt][r] * sc + sh;
        val *= mk[r];
        if (addres) val += (float)res[(rowbase + xr) * Cout + ch];
        if (relu) val = fmaxf(val, 0.0f);
        out[(rowbase + xr) * Cout + ch] = (__bf16)val;
      }
    }
  }
}

// ------------------------------------------------------------- host plumbing

namespace {

struct WsAlloc {
  char* base; size_t off; size_t cap;
  void* get(size_t bytes) {
    off = (off + 255) & ~(size_t)255;
    void* p = base + off;
    off += bytes;
    return p;
  }
};

struct ConvArgs {
  const __bf16 *in, *wp; const float* mask; const __bf16* res;
  const float *g, *b, *m, *v, *bias; __bf16* out;
  int Din, Hin, Win, Dout, Hout, Wout, Cin, Cout, kd, kh, kw;
  int sz, sy, sx, pz, py, px, relu, addres;
};

void launch_conv(const ConvArgs& a, int NT, int KC, hipStream_t s) {
  dim3 grid((unsigned)((a.Wout + 15) / 16), (unsigned)((a.Hout + 3) / 4),
            (unsigned)a.Dout);
  dim3 blk(128);
#define CALLC(nt, kc)                                                          \
  conv_wmma_kernel<nt, kc><<<grid, blk, 0, s>>>(                               \
      a.in, a.wp, a.mask, a.res, a.g, a.b, a.m, a.v, a.bias, a.out, a.Din,     \
      a.Hin, a.Win, a.Dout, a.Hout, a.Wout, a.Cin, a.Cout, a.kd, a.kh, a.kw,   \
      a.sz, a.sy, a.sx, a.pz, a.py, a.px, a.relu, a.addres)
  if      (NT == 1 && KC == 1) CALLC(1, 1);
  else if (NT == 2 && KC == 1) CALLC(2, 1);
  else if (NT == 4 && KC == 1) CALLC(4, 1);
  else if (NT == 4 && KC == 2) CALLC(4, 2);
  else if (NT == 8 && KC == 2) CALLC(8, 2);
  else if (NT == 8 && KC == 4) CALLC(8, 4);
#undef CALLC
}

void do_conv(void* const* d_in, WsAlloc& ws, hipStream_t s,
             int widx, int biasidx, const int* bn, int CinW,
             const __bf16* in, __bf16* out, const float* mask, const __bf16* res,
             int Din, int Hin, int Win, int Dout, int Hout, int Wout,
             int CinA, int Cout, int kd, int kh, int kw,
             int sz, int sy, int sx, int pz, int py, int px, int relu) {
  const int KC = (CinA + 31) / 32;
  const int NT = Cout / 16;
  const int taps = kd * kh * kw;
  const long pelems = (long)taps * KC * 2 * Cout * 16;
  __bf16* packed = (__bf16*)ws.get((size_t)pelems * sizeof(__bf16));
  repack_kernel<<<dim3((unsigned)((pelems + 255) / 256)), dim3(256), 0, s>>>(
      (const float*)d_in[widx], packed, Cout, CinW, taps, KC);
  ConvArgs a{in, packed, mask, res,
             (const float*)d_in[bn[0]], (const float*)d_in[bn[1]],
             (const float*)d_in[bn[2]], (const float*)d_in[bn[3]],
             biasidx >= 0 ? (const float*)d_in[biasidx] : nullptr, out,
             Din, Hin, Win, Dout, Hout, Wout, CinA, Cout, kd, kh, kw,
             sz, sy, sx, pz, py, px, relu, res ? 1 : 0};
  launch_conv(a, NT, KC, s);
}

struct BlkIdx { int w1, b1, bn1[4], w2, b2, bn2[4]; };
struct LvlIdx { int down_w; int down_bn[4]; BlkIdx blk[2];
                int tail_w, tail_bn[4]; int elk_w, elk_bn[4]; };

void run_level(void* const* d_in, WsAlloc& ws, hipStream_t s, const LvlIdx& L,
               __bf16* A, __bf16* B, __bf16* C, const float* mask,
               int D, int H, int W, int Cc) {
  auto same = [&](int widx, int biasidx, const int* bn, const __bf16* in,
                  __bf16* out, const __bf16* res, int relu) {
    do_conv(d_in, ws, s, widx, biasidx, bn, Cc, in, out, mask, res,
            D, H, W, D, H, W, Cc, Cc, 3, 3, 3, 1, 1, 1, 1, 1, 1, relu);
  };
  same(L.blk[0].w1, L.blk[0].b1, L.blk[0].bn1, A, B, nullptr, 1);
  same(L.blk[0].w2, L.blk[0].b2, L.blk[0].bn2, B, C, A, 1);       // +x, relu
  same(L.blk[1].w1, L.blk[1].b1, L.blk[1].bn1, C, B, nullptr, 1);
  same(L.blk[1].w2, L.blk[1].b2, L.blk[1].bn2, B, C, C, 1);       // in-place res
  same(L.tail_w, -1, L.tail_bn, C, B, nullptr, 0);                // bn*mask
  same(L.elk_w,  -1, L.elk_bn,  A, C, B, 1);                      // relu(h+lk)
}

} // namespace

extern "C" void kernel_launch(void* const* d_in, const int* in_sizes, int n_in,
                              void* d_out, int out_size, void* d_ws, size_t ws_size,
                              hipStream_t stream) {
  (void)n_in; (void)out_size;
  hipStream_t s = stream;

  // ---- fixed geometry (from reference: D = Dg+1 = 41, H = W = 96) ----
  const int D1 = 41, H1 = 96, W1 = 96; const long N1 = (long)D1 * H1 * W1;
  const int D2 = 21, H2 = 48, W2 = 48; const long N2 = (long)D2 * H2 * W2;
  const int D3 = 11, H3 = 24, W3 = 24; const long N3 = (long)D3 * H3 * W3;
  const int D4 = 5,  H4 = 12, W4 = 12; const long N4 = (long)D4 * H4 * W4;
  const int D5 = 2,  H5 = 12, W5 = 12; const long N5 = (long)D5 * H5 * W5;

  // ---- parameter index table (insertion-order flattening) ----
  int idx = 4;
  const int stem_w = idx++;
  int stem_bn[4]; for (int i = 0; i < 4; ++i) stem_bn[i] = idx++;
  LvlIdx L[4];
  for (int i = 0; i < 4; ++i) {
    if (i > 0) { L[i].down_w = idx++; for (int j = 0; j < 4; ++j) L[i].down_bn[j] = idx++; }
    else L[i].down_w = -1;
    for (int b = 0; b < 2; ++b) {
      L[i].blk[b].w1 = idx++; L[i].blk[b].b1 = idx++;
      for (int j = 0; j < 4; ++j) L[i].blk[b].bn1[j] = idx++;
      L[i].blk[b].w2 = idx++; L[i].blk[b].b2 = idx++;
      for (int j = 0; j < 4; ++j) L[i].blk[b].bn2[j] = idx++;
    }
    L[i].tail_w = idx++; for (int j = 0; j < 4; ++j) L[i].tail_bn[j] = idx++;
    L[i].elk_w  = idx++; for (int j = 0; j < 4; ++j) L[i].elk_bn[j]  = idx++;
  }
  const int extra_w = idx++;
  int extra_bn[4]; for (int i = 0; i < 4; ++i) extra_bn[i] = idx++;

  // ---- workspace layout (deterministic bump allocator) ----
  WsAlloc ws{(char*)d_ws, 0, ws_size};
  __bf16* IN = (__bf16*)ws.get((size_t)N1 * 16 * 2);   // stem input, 16-ch padded
  float*  m1 = (float*)ws.get((size_t)N1 * 4);
  __bf16* A1 = (__bf16*)ws.get((size_t)N1 * 16 * 2);
  __bf16* B1 = (__bf16*)ws.get((size_t)N1 * 16 * 2);
  __bf16* C1 = (__bf16*)ws.get((size_t)N1 * 16 * 2);
  float*  m2 = (float*)ws.get((size_t)N2 * 4);
  __bf16* A2 = (__bf16*)ws.get((size_t)N2 * 32 * 2);
  __bf16* B2 = (__bf16*)ws.get((size_t)N2 * 32 * 2);
  __bf16* C2 = (__bf16*)ws.get((size_t)N2 * 32 * 2);
  float*  m3 = (float*)ws.get((size_t)N3 * 4);
  __bf16* A3 = (__bf16*)ws.get((size_t)N3 * 64 * 2);
  __bf16* B3 = (__bf16*)ws.get((size_t)N3 * 64 * 2);
  __bf16* C3 = (__bf16*)ws.get((size_t)N3 * 64 * 2);
  float*  m4 = (float*)ws.get((size_t)N4 * 4);
  __bf16* A4 = (__bf16*)ws.get((size_t)N4 * 128 * 2);
  __bf16* B4 = (__bf16*)ws.get((size_t)N4 * 128 * 2);
  __bf16* C4 = (__bf16*)ws.get((size_t)N4 * 128 * 2);
  float*  m5 = (float*)ws.get((size_t)N5 * 4);
  __bf16* EX = (__bf16*)ws.get((size_t)N5 * 128 * 2);

  // ---- scatter sparse voxels into the dense grid ----
  const int NVOX = in_sizes[1] / 4;
  const int CIN  = NVOX > 0 ? in_sizes[0] / NVOX : 5;
  {
    long nw = N1 * 16 * 2 / 4;  // IN as u32 words
    fill_u32_kernel<<<dim3((unsigned)((nw + 255) / 256)), dim3(256), 0, s>>>(
        (unsigned int*)IN, 0u, nw);
    fill_u32_kernel<<<dim3((unsigned)((N1 + 255) / 256)), dim3(256), 0, s>>>(
        (unsigned int*)m1, 0u, N1);
    scatter_kernel<<<dim3((unsigned)((NVOX + 255) / 256)), dim3(256), 0, s>>>(
        (const float*)d_in[0], (const int*)d_in[1], IN, m1, NVOX, H1, W1, 16, CIN);
  }

  // ---- stem: dense(16-pad) -> 16ch, bn+mask+relu ----
  do_conv(d_in, ws, s, stem_w, -1, stem_bn, CIN, IN, A1, m1, nullptr,
          D1, H1, W1, D1, H1, W1, 16, 16, 3, 3, 3, 1, 1, 1, 1, 1, 1, 1);
  run_level(d_in, ws, s, L[0], A1, B1, C1, m1, D1, H1, W1, 16);

  // ---- level 2 (stride-2 down, 16->32) ----
  mask_down_kernel<<<dim3((unsigned)((N2 + 255) / 256)), dim3(256), 0, s>>>(
      m1, m2, D1, H1, W1, D2, H2, W2, 3, 3, 3, 2, 2, 2, 1, 1, 1);
  do_conv(d_in, ws, s, L[1].down_w, -1, L[1].down_bn, 16, C1, A2, m2, nullptr,
          D1, H1, W1, D2, H2, W2, 16, 32, 3, 3, 3, 2, 2, 2, 1, 1, 1, 1);
  run_level(d_in, ws, s, L[1], A2, B2, C2, m2, D2, H2, W2, 32);

  // ---- level 3 (32->64) ----
  mask_down_kernel<<<dim3((unsigned)((N3 + 255) / 256)), dim3(256), 0, s>>>(
      m2, m3, D2, H2, W2, D3, H3, W3, 3, 3, 3, 2, 2, 2, 1, 1, 1);
  do_conv(d_in, ws, s, L[2].down_w, -1, L[2].down_bn, 32, C2, A3, m3, nullptr,
          D2, H2, W2, D3, H3, W3, 32, 64, 3, 3, 3, 2, 2, 2, 1, 1, 1, 1);
  run_level(d_in, ws, s, L[2], A3, B3, C3, m3, D3, H3, W3, 64);

  // ---- level 4 (64->128, D-pad 0) ----
  mask_down_kernel<<<dim3((unsigned)((N4 + 255) / 256)), dim3(256), 0, s>>>(
      m3, m4, D3, H3, W3, D4, H4, W4, 3, 3, 3, 2, 2, 2, 0, 1, 1);
  do_conv(d_in, ws, s, L[3].down_w, -1, L[3].down_bn, 64, C3, A4, m4, nullptr,
          D3, H3, W3, D4, H4, W4, 64, 128, 3, 3, 3, 2, 2, 2, 0, 1, 1, 1);
  run_level(d_in, ws, s, L[3], A4, B4, C4, m4, D4, H4, W4, 128);

  // ---- extra (3,1,1) stride (2,1,1), no pad ----
  mask_down_kernel<<<dim3((unsigned)((N5 + 255) / 256)), dim3(256), 0, s>>>(
      m4, m5, D4, H4, W4, D5, H5, W5, 3, 1, 1, 2, 1, 1, 0, 0, 0);
  do_conv(d_in, ws, s, extra_w, -1, extra_bn, 128, C4, EX, m5, nullptr,
          D4, H4, W4, D5, H5, W5, 128, 128, 3, 1, 1, 2, 1, 1, 0, 0, 0, 1);

  // ---- reshape (C,D,H,W)->(C*D,H,W) fp32 ----
  {
    int total = 128 * D5 * H5 * W5;
    final_kernel<<<dim3((unsigned)((total + 255) / 256)), dim3(256), 0, s>>>(
        EX, (float*)d_out, D5, H5, W5, 128);
  }
}
